// LocalGQAttention_69011534512409
// MI455X (gfx1250) — matrix operand order repaired and checked by reading, not compiled, not run on hardware
//
#include <hip/hip_runtime.h>
#include <math.h>
#include <stdint.h>

#define T_SEQ   4096
#define HIDDEN  2048
#define QH      16
#define KVH     4
#define HD      128
#define WIN     512
#define KDIM    (KVH * HD)   /* 512 */

typedef __attribute__((ext_vector_type(16))) __bf16 v16bf;
typedef __attribute__((ext_vector_type(8)))  __bf16 v8bf;
typedef __attribute__((ext_vector_type(8)))  float  v8f;
typedef __attribute__((ext_vector_type(4)))  unsigned int v4u;
typedef __attribute__((ext_vector_type(8)))  int v8i;
typedef __attribute__((ext_vector_type(4)))  int v4i;

// ---------------------------------------------------------------------------
// WMMA fragment helpers (CDNA5 wave32 layouts, ISA 7.12.2)
// A (16x32 bf16): lane row = lane&15; chunks K = (lane>>4)*8 .. +7 and +16..+23
// B (32x16 bf16): lane col = lane&15; K chunks (lane>>4)*8 .. +7 and +16..+23, as A
// C/D (16x16 f32): lane col = lane&15; vgpr i -> row i + (lane>>4)*8
// ---------------------------------------------------------------------------
__device__ __forceinline__ v16bf frag_a(const __bf16* base, int ld) {
  const int lane = threadIdx.x & 31;
  const __bf16* pr = base + (size_t)(lane & 15) * ld + ((lane >> 4) << 3);
  union { v16bf v; v8bf h[2]; } u;
  u.h[0] = *reinterpret_cast<const v8bf*>(pr);
  u.h[1] = *reinterpret_cast<const v8bf*>(pr + 16);
  return u.v;
}

__device__ __forceinline__ v16bf frag_b(const __bf16* base, int ld) {
  const int lane = threadIdx.x & 31;
  const __bf16* pr = base + (size_t)(lane & 15) * ld + ((lane >> 4) << 3);
  union { v16bf v; v8bf h[2]; } u;
  u.h[0] = *reinterpret_cast<const v8bf*>(pr);
  u.h[1] = *reinterpret_cast<const v8bf*>(pr + 16);
  return u.v;
}

__device__ __forceinline__ v8f wmma_bf16(v16bf a, v16bf b, v8f c) {
  return __builtin_amdgcn_wmma_f32_16x16x32_bf16(
      /*neg_a=*/false, a, /*neg_b=*/false, b,
      /*c_mod=*/(short)0, c, /*reuse_a=*/false, /*reuse_b=*/false);
}

// ---------------------------------------------------------------------------
// Tensor Data Mover: 2-D tile load, global -> LDS (ISA 08_async_tensor §8)
//   group0: [1:0]=count=1, [63:32]=lds byte addr, [120:64]=global addr, type=2
//   group1: data_size=1 (2B), tile_dim0=rowlen, tile_dim1=rows,
//           tensor_dim0/1 large (no clipping), tensor_dim0_stride = ld (elems)
//   groups 2/3 + trailing group: zero (2-D tile)
// 6-arg builtin on this toolchain: (v4u, v8i, v4i, v4i, v8i, i32 cpol)
// ---------------------------------------------------------------------------
__device__ __forceinline__ void tdm_load_2d(unsigned int lds_off, const void* gaddr,
                                            unsigned int rows, unsigned int rowlen,
                                            unsigned long long stride_elts) {
  const unsigned long long ga = (unsigned long long)(uintptr_t)gaddr;
  const unsigned int td0 = 1u << 20, td1 = 1u << 20;   // generous tensor dims
  v4u g0;
  g0.x = 1u;                                            // count = 1, user mode
  g0.y = lds_off;                                       // LDS byte address
  g0.z = (unsigned int)(ga & 0xFFFFFFFFu);              // global addr [31:0]
  g0.w = (unsigned int)((ga >> 32) & 0x01FFFFFFu) | (2u << 30);  // [56:32] | type=2
  v8i g1;
  g1[0] = (int)(1u << 16);                              // data_size = 1 -> 2 bytes
  g1[1] = (int)((td0 & 0xFFFFu) << 16);                 // tensor_dim0 [15:0]
  g1[2] = (int)(((td0 >> 16) & 0xFFFFu) | ((td1 & 0xFFFFu) << 16));
  g1[3] = (int)(((td1 >> 16) & 0xFFFFu) | ((rowlen & 0xFFFFu) << 16)); // tile_dim0
  g1[4] = (int)(rows & 0xFFFFu);                        // tile_dim1 (tile_dim2=0)
  g1[5] = (int)(stride_elts & 0xFFFFFFFFull);           // dim0_stride [31:0]
  g1[6] = (int)((stride_elts >> 32) & 0xFFFFull);       // dim0_stride [47:32]
  g1[7] = 0;                                            // dim1_stride = 0
  v4i g2 = {0, 0, 0, 0};
  v4i g3 = {0, 0, 0, 0};
  v8i g4 = {0, 0, 0, 0, 0, 0, 0, 0};
  __builtin_amdgcn_tensor_load_to_lds(g0, g1, g2, g3, g4, 0);
}

// ---------------------------------------------------------------------------
// fp32 -> bf16 elementwise convert
// ---------------------------------------------------------------------------
__global__ void cvt_bf16_kernel(const float* __restrict__ in,
                                __bf16* __restrict__ out, size_t n) {
  size_t i = (size_t)blockIdx.x * blockDim.x + threadIdx.x;
  if (i < n) out[i] = (__bf16)in[i];
}

// ---------------------------------------------------------------------------
// TDM-staged bf16 WMMA GEMM:  C[M,N] (f32) = A[M,K] * B[N,K]^T
// block = 256 threads (8 waves); wave tile 32x32; workgroup tile 64x128.
// K consumed in 64-wide blocks, double-buffered in LDS via tensor_load_to_lds.
// grid.x = M/64, grid.y = N/128; K multiple of 64.
// LDS: A 2x(64x64) bf16 @ [0,16K), B 2x(128x64) bf16 @ [16K,48K).
// ---------------------------------------------------------------------------
#define KB 64
__global__ __launch_bounds__(256)
void gemm_bf16_kernel(const __bf16* __restrict__ A, const __bf16* __restrict__ B,
                      float* __restrict__ C, int K, int lda, int ldb, int ldc) {
  __shared__ __bf16 smem[2 * 64 * KB + 2 * 128 * KB];   // 48 KB
  const int lane = threadIdx.x & 31;
  const int wave = threadIdx.x >> 5;
  const int wvM  = wave >> 2;           // 0..1
  const int wvN  = wave & 3;            // 0..3
  const int tM   = blockIdx.x * 64 + wvM * 32;
  const int tN   = blockIdx.y * 128 + wvN * 32;
  const unsigned int ldsBase = (unsigned int)(uintptr_t)(void*)&smem[0];

  const __bf16* Ag = A + (size_t)(blockIdx.x * 64) * lda;
  const __bf16* Bg = B + (size_t)(blockIdx.y * 128) * ldb;

  v8f c00 = {}, c01 = {}, c10 = {}, c11 = {};

  const int nStages = K / KB;
  if (wave == 0) {
    tdm_load_2d(ldsBase,          Ag,  64, KB, (unsigned long long)lda);
    tdm_load_2d(ldsBase + 16384,  Bg, 128, KB, (unsigned long long)ldb);
  }
  int buf = 0;
  for (int st = 0; st < nStages; ++st) {
    if (wave == 0) __builtin_amdgcn_s_wait_tensorcnt(0);  // current buf landed
    __syncthreads();                                      // publish to all waves
    if (wave == 0 && st + 1 < nStages) {                  // prefetch next block
      const int k0 = (st + 1) * KB;
      tdm_load_2d(ldsBase + (unsigned)(buf ^ 1) * 8192,          Ag + k0,  64, KB,
                  (unsigned long long)lda);
      tdm_load_2d(ldsBase + 16384 + (unsigned)(buf ^ 1) * 16384, Bg + k0, 128, KB,
                  (unsigned long long)ldb);
    }
    const __bf16* Abuf = smem + (size_t)buf * (64 * KB);
    const __bf16* Bbuf = smem + 2 * 64 * KB + (size_t)buf * (128 * KB);
    const __bf16* Aw0 = Abuf + (size_t)(wvM * 32) * KB;
    const __bf16* Aw1 = Abuf + (size_t)(wvM * 32 + 16) * KB;
    const __bf16* Bw0 = Bbuf + (size_t)(wvN * 32) * KB;
    const __bf16* Bw1 = Bbuf + (size_t)(wvN * 32 + 16) * KB;
#pragma unroll
    for (int kk = 0; kk < KB; kk += 32) {
      v16bf a0 = frag_a(Aw0 + kk, KB);
      v16bf a1 = frag_a(Aw1 + kk, KB);
      v16bf b0 = frag_b(Bw0 + kk, KB);
      v16bf b1 = frag_b(Bw1 + kk, KB);
      c00 = wmma_bf16(a0, b0, c00);
      c01 = wmma_bf16(a0, b1, c01);
      c10 = wmma_bf16(a1, b0, c10);
      c11 = wmma_bf16(a1, b1, c11);
    }
    __syncthreads();                 // everyone done reading buf
    buf ^= 1;
  }

  const int col = lane & 15;
  const int rb  = (lane >> 4) << 3;
  float* Cp0 = C + (size_t)(tM + rb) * ldc + tN + col;
  float* Cp1 = C + (size_t)(tM + 16 + rb) * ldc + tN + col;
#pragma unroll
  for (int i = 0; i < 8; ++i) {
    Cp0[(size_t)i * ldc]      = c00[i];
    Cp0[(size_t)i * ldc + 16] = c01[i];
    Cp1[(size_t)i * ldc]      = c10[i];
    Cp1[(size_t)i * ldc + 16] = c11[i];
  }
}

// ---------------------------------------------------------------------------
// RoPE + cast to bf16. One thread per (t, head, d<64) rotation pair.
// ---------------------------------------------------------------------------
__global__ void rope_kernel(const float* __restrict__ in, __bf16* __restrict__ out,
                            int H, int ld, int total) {
  int idx = blockIdx.x * blockDim.x + threadIdx.x;
  if (idx >= total) return;
  const int d = idx & 63;
  const int h = (idx >> 6) % H;
  const int t = idx / (64 * H);
  const size_t off = (size_t)t * ld + (size_t)h * HD;
  // inv_freq = 10000^(-d/64) = exp2(-d * log2(10000)/64)
  const float inv = __builtin_exp2f(-(float)d * 0.20762050593046015f);
  const float ang = (float)t * inv;
  float s, c;
  sincosf(ang, &s, &c);
  const float x1 = in[off + d];
  const float x2 = in[off + d + 64];
  out[off + d]      = (__bf16)(x1 * c - x2 * s);
  out[off + d + 64] = (__bf16)(x1 * s + x2 * c);
}

// V transpose: vT[kvh*128 + d][t] = v[t][kvh*128 + d]  (bf16 cast)
__global__ void vtrans_kernel(const float* __restrict__ vf, __bf16* __restrict__ vT) {
  int idx = blockIdx.x * blockDim.x + threadIdx.x;
  if (idx >= T_SEQ * KDIM) return;
  const int t = idx >> 9;      // / 512
  const int c = idx & 511;
  vT[(size_t)c * T_SEQ + t] = (__bf16)vf[idx];
}

// ---------------------------------------------------------------------------
// Flash-style sliding-window GQA attention.
// One wave = one (16-row query tile, q-head). block = 128 (4 waves).
// grid.x = (T/16)*QH/4 = 1024.
// ---------------------------------------------------------------------------
__global__ __launch_bounds__(128)
void attn_kernel(const __bf16* __restrict__ qb, const __bf16* __restrict__ kb,
                 const __bf16* __restrict__ vT, __bf16* __restrict__ attnb) {
  __shared__ __bf16 plds[4][16 * 32];   // per-wave P tile, A-layout staging
  const int lane = threadIdx.x & 31;
  const int wave = threadIdx.x >> 5;
  const int task = blockIdx.x * 4 + wave;     // 0..4095
  const int qh   = task & (QH - 1);
  const int qs   = (task >> 4) * 16;          // query tile start
  const int kvh  = qh >> 2;

  const __bf16* Qb = qb + (size_t)qs * HIDDEN + (size_t)qh * HD;
  const __bf16* Kb = kb + (size_t)kvh * HD;
  const __bf16* Vb = vT + (size_t)kvh * HD * T_SEQ;
  __bf16* Pl = &plds[wave][0];

  // Q fragments for the whole 128-dim head (4 k-steps of 32)
  v16bf qf[4];
#pragma unroll
  for (int kk = 0; kk < 4; ++kk) qf[kk] = frag_a(Qb + kk * 32, HIDDEN);

  v8f o[8];
#pragma unroll
  for (int d = 0; d < 8; ++d) o[d] = (v8f){};

  float mrow[8], lrow[8];
#pragma unroll
  for (int i = 0; i < 8; ++i) { mrow[i] = -3.0e38f; lrow[i] = 0.0f; }

  const int mAdd = (lane >> 4) << 3;   // row offset of this lane-half in C layout
  const int nCol = lane & 15;
  const float sscale = 0.12752409768829023f;  // (1/sqrt(128)) * log2(e)

  int s_begin = qs - WIN; if (s_begin < 0) s_begin = 0; s_begin &= ~31;
  const int s_end = qs + 16;   // keys s <= qs+15; max touched row stays < T

  for (int s0 = s_begin; s0 < s_end; s0 += 32) {
    // ---- S = Q * K^T for a 16x32 key block (two 16-wide C frags) ----
    v8f sf0 = {}, sf1 = {};
#pragma unroll
    for (int kk = 0; kk < 4; ++kk) {
      v16bf kf0 = frag_b(Kb + (size_t)s0 * KDIM + kk * 32, KDIM);
      v16bf kf1 = frag_b(Kb + (size_t)(s0 + 16) * KDIM + kk * 32, KDIM);
      sf0 = wmma_bf16(qf[kk], kf0, sf0);
      sf1 = wmma_bf16(qf[kk], kf1, sf1);
    }

    // ---- mask + online softmax (row stats across 16-lane halves) ----
#pragma unroll
    for (int i = 0; i < 8; ++i) {
      const int t  = qs + mAdd + i;
      const int sA = s0 + nCol;
      const int sB = sA + 16;
      const bool okA = (sA <= t) && ((t - sA) <= WIN);
      const bool okB = (sB <= t) && ((t - sB) <= WIN);
      float zA = okA ? sf0[i] * sscale : -3.0e38f;
      float zB = okB ? sf1[i] * sscale : -3.0e38f;
      float mx = fmaxf(zA, zB);
#pragma unroll
      for (int off = 1; off < 16; off <<= 1)
        mx = fmaxf(mx, __shfl_xor(mx, off, 32));
      const float mnew = fmaxf(mrow[i], mx);
      const float corr = __builtin_exp2f(mrow[i] - mnew);
      const float pA = okA ? __builtin_exp2f(zA - mnew) : 0.0f;
      const float pB = okB ? __builtin_exp2f(zB - mnew) : 0.0f;
      float rs = pA + pB;
#pragma unroll
      for (int off = 1; off < 16; off <<= 1)
        rs += __shfl_xor(rs, off, 32);
      lrow[i] = lrow[i] * corr + rs;
      mrow[i] = mnew;
#pragma unroll
      for (int d = 0; d < 8; ++d) o[d][i] = o[d][i] * corr;
      const int m = mAdd + i;                 // row in P tile
      Pl[m * 32 + nCol]      = (__bf16)pA;
      Pl[m * 32 + 16 + nCol] = (__bf16)pB;
    }

    // in-wave LDS store->load ordering
    asm volatile("s_wait_dscnt 0" ::: "memory");

    // ---- reload P as an A fragment (16x32) ----
    v16bf pf;
    {
      const __bf16* pp = Pl + nCol * 32 + mAdd;
      union { v16bf v; v8bf h[2]; } u;
      u.h[0] = *reinterpret_cast<const v8bf*>(pp);
      u.h[1] = *reinterpret_cast<const v8bf*>(pp + 16);
      pf = u.v;
    }

    // ---- O += P * V (V^T rows are s-contiguous -> B fragments) ----
#pragma unroll
    for (int dt = 0; dt < 8; ++dt) {
      v16bf vfr = frag_b(Vb + (size_t)(dt * 16) * T_SEQ + s0, T_SEQ);
      o[dt] = wmma_bf16(pf, vfr, o[dt]);
    }
  }

  // ---- normalize and write attn output (bf16) ----
#pragma unroll
  for (int i = 0; i < 8; ++i) {
    const float inv = 1.0f / lrow[i];
    const int t = qs + mAdd + i;
    __bf16* orow = attnb + (size_t)t * HIDDEN + (size_t)qh * HD + nCol;
#pragma unroll
    for (int dt = 0; dt < 8; ++dt)
      orow[dt * 16] = (__bf16)(o[dt][i] * inv);
  }
}

// ---------------------------------------------------------------------------
// Host launcher
// ---------------------------------------------------------------------------
extern "C" void kernel_launch(void* const* d_in, const int* in_sizes, int n_in,
                              void* d_out, int out_size, void* d_ws, size_t ws_size,
                              hipStream_t stream) {
  const float* x  = (const float*)d_in[0];
  const float* wq = (const float*)d_in[1];
  const float* wk = (const float*)d_in[2];
  const float* wv = (const float*)d_in[3];
  const float* wo = (const float*)d_in[4];
  float* out = (float*)d_out;

  char* p = (char*)d_ws;
  auto take = [&p](size_t bytes) -> char* {
    char* r = p;
    p += (bytes + 255) & ~(size_t)255;
    return r;
  };
  __bf16* xb    = (__bf16*)take((size_t)T_SEQ * HIDDEN * 2);
  __bf16* wqb   = (__bf16*)take((size_t)HIDDEN * HIDDEN * 2);
  __bf16* wkb   = (__bf16*)take((size_t)KDIM * HIDDEN * 2);
  __bf16* wvb   = (__bf16*)take((size_t)KDIM * HIDDEN * 2);
  __bf16* wob   = (__bf16*)take((size_t)HIDDEN * HIDDEN * 2);
  float*  qf32  = (float*) take((size_t)T_SEQ * HIDDEN * 4);
  float*  kf32  = (float*) take((size_t)T_SEQ * KDIM * 4);
  float*  vf32  = (float*) take((size_t)T_SEQ * KDIM * 4);
  __bf16* qb    = (__bf16*)take((size_t)T_SEQ * HIDDEN * 2);
  __bf16* kb    = (__bf16*)take((size_t)T_SEQ * KDIM * 2);
  __bf16* vT    = (__bf16*)take((size_t)KDIM * T_SEQ * 2);
  __bf16* attnb = (__bf16*)take((size_t)T_SEQ * HIDDEN * 2);

  auto cvt = [&](const float* in, __bf16* o_, size_t n) {
    cvt_bf16_kernel<<<(unsigned)((n + 255) / 256), 256, 0, stream>>>(in, o_, n);
  };
  cvt(x,  xb,  (size_t)T_SEQ * HIDDEN);
  cvt(wq, wqb, (size_t)HIDDEN * HIDDEN);
  cvt(wk, wkb, (size_t)KDIM * HIDDEN);
  cvt(wv, wvb, (size_t)KDIM * HIDDEN);
  cvt(wo, wob, (size_t)HIDDEN * HIDDEN);

  // Q/K/V projections: C = x * w^T
  gemm_bf16_kernel<<<dim3(T_SEQ / 64, HIDDEN / 128), 256, 0, stream>>>(
      xb, wqb, qf32, HIDDEN, HIDDEN, HIDDEN, HIDDEN);
  gemm_bf16_kernel<<<dim3(T_SEQ / 64, KDIM / 128), 256, 0, stream>>>(
      xb, wkb, kf32, HIDDEN, HIDDEN, HIDDEN, KDIM);
  gemm_bf16_kernel<<<dim3(T_SEQ / 64, KDIM / 128), 256, 0, stream>>>(
      xb, wvb, vf32, HIDDEN, HIDDEN, HIDDEN, KDIM);

  // RoPE + casts, V transpose
  {
    int total = T_SEQ * QH * 64;
    rope_kernel<<<(total + 255) / 256, 256, 0, stream>>>(qf32, qb, QH, HIDDEN, total);
    total = T_SEQ * KVH * 64;
    rope_kernel<<<(total + 255) / 256, 256, 0, stream>>>(kf32, kb, KVH, KDIM, total);
    total = T_SEQ * KDIM;
    vtrans_kernel<<<(total + 255) / 256, 256, 0, stream>>>(vf32, vT);
  }

  // Attention
  attn_kernel<<<(T_SEQ / 16) * QH / 4, 128, 0, stream>>>(qb, kb, vT, attnb);

  // Output projection straight into d_out (f32)
  gemm_bf16_kernel<<<dim3(T_SEQ / 64, HIDDEN / 128), 256, 0, stream>>>(
      attnb, wob, out, HIDDEN, HIDDEN, HIDDEN, HIDDEN);
}
